// Top1Gate_53352083751353
// MI455X (gfx1250) — matrix operand order repaired and checked by reading, not compiled, not run on hardware
//
#include <hip/hip_runtime.h>
#include <hip/hip_bf16.h>
#include <stdint.h>

#define T_TOK 16384
#define D_DIM 4096
#define E_EXP 64

typedef __attribute__((ext_vector_type(16))) __bf16   v16bf;
typedef __attribute__((ext_vector_type(8)))  float    v8f;
typedef __attribute__((ext_vector_type(8)))  uint32_t v8u;

// ---------------------------------------------------------------------------
// Kernel 1: split wg (fp32 [E,D]) into bf16 hi/lo planes; zero accumulators.
// ---------------------------------------------------------------------------
__global__ void wg_split_kernel(const float* __restrict__ wg,
                                uint16_t* __restrict__ wg_hi,
                                uint16_t* __restrict__ wg_lo,
                                float* __restrict__ accum /*65 floats*/) {
  int i = blockIdx.x * blockDim.x + threadIdx.x;
  if (blockIdx.x == 0 && threadIdx.x < 65) accum[threadIdx.x] = 0.0f;
  if (i < E_EXP * D_DIM) {
    float f = wg[i];
    uint32_t u = __builtin_bit_cast(uint32_t, f);
    wg_hi[i] = (uint16_t)(u >> 16);  // truncated bf16 (exact prefix of fp32)
    float hf = __builtin_bit_cast(float, (uint32_t)(u & 0xFFFF0000u));
    __bf16 lo = (__bf16)(f - hf);    // hardware RNE cvt
    wg_lo[i] = __builtin_bit_cast(uint16_t, lo);
  }
}

// ---------------------------------------------------------------------------
// Kernel 2: split-precision bf16 WMMA GEMM (logits = x @ wg^T) + softmax,
// argmax, entropy, gate extraction. One wave handles 16 tokens x 64 experts.
// ---------------------------------------------------------------------------
__global__ void __launch_bounds__(128)
top1_gate_main(const float* __restrict__ x,
               const uint16_t* __restrict__ wg_hi,
               const uint16_t* __restrict__ wg_lo,
               float* __restrict__ out,
               float* __restrict__ me_sum,
               float* __restrict__ ent_sum) {
  const int wave = threadIdx.x >> 5;
  const int lane = threadIdx.x & 31;
  const int half = lane >> 4;   // which 16-lane half of the wave
  const int m    = lane & 15;   // row (A) / column (B,C,D) index
  const int tb   = blockIdx.x * 4 + wave;
  const int t0   = tb * 16;

  v8f acc[4] = { {}, {}, {}, {} };   // 16 tokens x 4 expert tiles of 16

  const float4* xr = (const float4*)(x + (size_t)(t0 + m) * D_DIM);

  // K loop: 32 elements of D per step, 128 steps.
  #pragma unroll 2
  for (int k4 = 0; k4 < D_DIM / 4; k4 += 8) {   // k4 in float4 units; k0 = k4*4
    // A fragment: lanes 0-15 hold K = 0..15, lanes 16-31 hold
    // K = 16..31 (relative to k0), row M = m. One 64B run per lane.
    float4 f0 = xr[k4 + half * 4];
    float4 f1 = xr[k4 + half * 4 + 1];
    float4 f2 = xr[k4 + half * 4 + 2];
    float4 f3 = xr[k4 + half * 4 + 3];
    float fa[16] = { f0.x, f0.y, f0.z, f0.w,  f1.x, f1.y, f1.z, f1.w,
                     f2.x, f2.y, f2.z, f2.w,  f3.x, f3.y, f3.z, f3.w };
    v8u   ahw;        // hi plane: 8 packed dwords (2 bf16 each)
    v16bf a_lo;       // lo plane
    #pragma unroll
    for (int p = 0; p < 8; ++p) {
      uint32_t u0 = __builtin_bit_cast(uint32_t, fa[2 * p]);
      uint32_t u1 = __builtin_bit_cast(uint32_t, fa[2 * p + 1]);
      // pack {trunc_bf16(u1), trunc_bf16(u0)} with one v_perm_b32
      ahw[p] = __builtin_amdgcn_perm(u1, u0, 0x07060302u);
      float h0 = __builtin_bit_cast(float, u0 & 0xFFFF0000u);
      float h1 = __builtin_bit_cast(float, u1 & 0xFFFF0000u);
      a_lo[2 * p]     = (__bf16)(fa[2 * p]     - h0);  // hardware RNE cvt
      a_lo[2 * p + 1] = (__bf16)(fa[2 * p + 1] - h1);
    }
    v16bf a_hi = __builtin_bit_cast(v16bf, ahw);

    const int k0 = k4 * 4;
    // B fragment: lane holds column N = m; lanes 0-15 K=0..15, 16-31 K=16..31.
    #pragma unroll
    for (int et = 0; et < 4; ++et) {
      const size_t boff = (size_t)(et * 16 + m) * D_DIM + (size_t)(k0 + half * 16);
      v16bf b_hi = __builtin_bit_cast(v16bf, *(const v8u*)(wg_hi + boff));
      v16bf b_lo = __builtin_bit_cast(v16bf, *(const v8u*)(wg_lo + boff));
      // fp32-ish accuracy: x_hi*w_hi + x_lo*w_hi + x_hi*w_lo
      acc[et] = __builtin_amdgcn_wmma_f32_16x16x32_bf16(
          false, a_hi, false, b_hi, (short)0, acc[et], false, false);
      acc[et] = __builtin_amdgcn_wmma_f32_16x16x32_bf16(
          false, a_lo, false, b_hi, (short)0, acc[et], false, false);
      acc[et] = __builtin_amdgcn_wmma_f32_16x16x32_bf16(
          false, a_hi, false, b_lo, (short)0, acc[et], false, false);
    }
  }

  // Epilogue. C layout: acc[et][r] = logits[t0 + half*8 + r][et*16 + m].
  // Each row's 64 expert values live across the 16 lanes of one half-group
  // (4 regs per lane) -> reduce with shfl_xor inside width-16 groups.
  float me_local[4] = { 0.f, 0.f, 0.f, 0.f };
  float ent_acc = 0.f;

  #pragma unroll
  for (int r = 0; r < 8; ++r) {
    float v0 = acc[0][r], v1 = acc[1][r], v2 = acc[2][r], v3 = acc[3][r];
    float lm = v0; int ai = m;
    if (v1 > lm) { lm = v1; ai = 16 + m; }
    if (v2 > lm) { lm = v2; ai = 32 + m; }
    if (v3 > lm) { lm = v3; ai = 48 + m; }
    #pragma unroll
    for (int s = 1; s < 16; s <<= 1) {
      float om = __shfl_xor(lm, s, 16);
      int   oi = __shfl_xor(ai, s, 16);
      if (om > lm || (om == lm && oi < ai)) { lm = om; ai = oi; }
    }
    float e0 = __expf(v0 - lm), e1 = __expf(v1 - lm);
    float e2 = __expf(v2 - lm), e3 = __expf(v3 - lm);
    float s_loc = e0 + e1 + e2 + e3;
    float w_loc = e0 * (v0 - lm) + e1 * (v1 - lm) + e2 * (v2 - lm) + e3 * (v3 - lm);
    #pragma unroll
    for (int s = 1; s < 16; s <<= 1) {
      s_loc += __shfl_xor(s_loc, s, 16);
      w_loc += __shfl_xor(w_loc, s, 16);
    }
    float inv = 1.0f / s_loc;
    me_local[0] += e0 * inv; me_local[1] += e1 * inv;
    me_local[2] += e2 * inv; me_local[3] += e3 * inv;
    if (m == 0) {
      int t = t0 + half * 8 + r;
      out[1 + t]          = inv;         // gate of argmax = exp(0)/S = 1/S
      out[1 + T_TOK + t]  = (float)ai;   // indices1_s (exact small int in fp32)
      ent_acc += __logf(s_loc) - w_loc * inv;  // -sum g log g
    }
  }

  #pragma unroll
  for (int et = 0; et < 4; ++et) {
    me_local[et] += __shfl_xor(me_local[et], 16, 32);  // fold the two halves
    if (half == 0) atomicAdd(&me_sum[et * 16 + m], me_local[et]);
  }
  if (m == 0) atomicAdd(ent_sum, ent_acc);
}

// ---------------------------------------------------------------------------
// Kernel 3: order-dependent per-expert queue positions (cumsum of one-hot),
// plus l_aux and entropy_mean finalization. Single 1024-thread block.
// ---------------------------------------------------------------------------
__global__ void __launch_bounds__(1024)
top1_locations_finalize(float* __restrict__ out,
                        const float* __restrict__ me_sum,
                        const float* __restrict__ ent_sum) {
  __shared__ int   counts[32][64];
  __shared__ int   prefixS[32][64];
  __shared__ int   baseS[64];
  __shared__ float red[64];
  const int tid = threadIdx.x;
  const int wv  = tid >> 5;
  const int ln  = tid & 31;
  if (tid < 64) baseS[tid] = 0;
  __syncthreads();

  const float* idxf = out + 1 + T_TOK;
  float*       locf = out + 1 + 2 * T_TOK;

  for (int c = 0; c < T_TOK / 1024; ++c) {
    int t = c * 1024 + tid;
    int e = (int)idxf[t];
    // match_any over 6 expert-index bits -> mask of lanes with same expert
    unsigned mask = 0xFFFFFFFFu;
    #pragma unroll
    for (int b = 0; b < 6; ++b) {
      bool mine = (e >> b) & 1;
      unsigned bal = (unsigned)__ballot(mine);
      mask &= mine ? bal : ~bal;
    }
    int rank = __popc(mask & ((1u << ln) - 1u));
    int* cflat = &counts[0][0];
    cflat[tid] = 0; cflat[tid + 1024] = 0;
    __syncthreads();
    if (rank == 0) counts[wv][e] = __popc(mask);
    __syncthreads();
    if (tid < 64) {                      // exclusive prefix over the 32 waves
      int run = baseS[tid];
      for (int w = 0; w < 32; ++w) { prefixS[w][tid] = run; run += counts[w][tid]; }
      baseS[tid] = run;                  // running per-expert base across chunks
    }
    __syncthreads();
    locf[t] = (float)(prefixS[wv][e] + rank);
  }
  __syncthreads();

  if (tid < 64) {
    const float invT = 1.0f / (float)T_TOK;
    red[tid] = (me_sum[tid] * invT) * ((float)baseS[tid] * invT);  // me*ce
  }
  __syncthreads();
  if (tid == 0) {
    float s = 0.f;
    for (int e = 0; e < 64; ++e) s += red[e];
    out[0] = s * 64.0f;                           // mean(me*ce)*E*E = sum*64
    out[1 + 3 * T_TOK] = ent_sum[0] * (1.0f / (float)T_TOK);
  }
}

// ---------------------------------------------------------------------------
extern "C" void kernel_launch(void* const* d_in, const int* in_sizes, int n_in,
                              void* d_out, int out_size, void* d_ws, size_t ws_size,
                              hipStream_t stream) {
  const float* x  = (const float*)d_in[0];   // [T, D] fp32
  const float* wg = (const float*)d_in[1];   // [E, D] fp32

  uint16_t* wg_hi = (uint16_t*)d_ws;
  uint16_t* wg_lo = wg_hi + (size_t)E_EXP * D_DIM;
  float* accum    = (float*)((char*)d_ws + 2 * (size_t)E_EXP * D_DIM * sizeof(uint16_t));
  float* me_sum   = accum;        // 64 floats
  float* ent_sum  = accum + 64;   // 1 float
  float* out      = (float*)d_out;

  wg_split_kernel<<<(E_EXP * D_DIM + 255) / 256, 256, 0, stream>>>(
      wg, wg_hi, wg_lo, accum);
  top1_gate_main<<<T_TOK / 16 / 4, 128, 0, stream>>>(
      x, wg_hi, wg_lo, out, me_sum, ent_sum);
  top1_locations_finalize<<<1, 1024, 0, stream>>>(out, me_sum, ent_sum);
}
